// CARNN_78950088835708
// MI455X (gfx1250) — compile-verified, matrix operand order checked by reading, not run on hardware
//
#include <hip/hip_runtime.h>
#include <math.h>

typedef __attribute__((ext_vector_type(16))) _Float16 v16h;
typedef __attribute__((ext_vector_type(8)))  _Float16 v8h;
typedef __attribute__((ext_vector_type(4)))  _Float16 v4h;
typedef __attribute__((ext_vector_type(8)))  float    v8f;
typedef __attribute__((ext_vector_type(4)))  float    v4f;

#define TM 128
#define TN 128
#define TK 32
#define LDR 40   // LDS row stride in halfs: 80B rows -> 16B-aligned fragment loads

// One 16x32 f16 WMMA operand fragment per lane from an LDS tile row.
// ISA 7.12.2 (16-bit A 16x32): lane -> row lane%16, half h=lane/16;
// elems 0..7 -> K=8h..8h+7, elems 8..15 -> K=16+8h..16+8h+7 (two b128 reads).
__device__ __forceinline__ v16h frag_ld(const _Float16* rowp, int h) {
  v8h lo = *reinterpret_cast<const v8h*>(rowp + 8 * h);
  v8h hi = *reinterpret_cast<const v8h*>(rowp + 16 + 8 * h);
  v16h f;
#pragma unroll
  for (int i = 0; i < 8; ++i) { f[i] = lo[i]; f[8 + i] = hi[i]; }
  return f;
}

// ---- register-staged tile loaders: clamp+select (no EXEC branches) ----------
__device__ __forceinline__ void loadA_f32(const float* __restrict__ A, int M, int K,
                                          int m0, int k0, int tid, v4f* ar) {
  const int c4 = (tid & 7) * 4;
  const bool full = (m0 + TM <= M) && (k0 + TK <= K) && ((K & 3) == 0);
#pragma unroll
  for (int i = 0; i < 4; ++i) {
    int r = (tid >> 3) + 32 * i;
    int m = m0 + r, k = k0 + c4;
    if (full) {
      ar[i] = *(const v4f*)(A + (size_t)m * K + k);
    } else {
      int mc = m < M ? m : M - 1;
#pragma unroll
      for (int j = 0; j < 4; ++j) {
        int kc = (k + j) < K ? (k + j) : K - 1;
        float v = A[(size_t)mc * K + kc];
        ar[i][j] = (m < M && (k + j) < K) ? v : 0.f;
      }
    }
  }
}

// B is pre-converted f16 [K][N]; each thread owns 4 adjacent K rows x 4 columns
// -> register transpose -> 4 aligned b64 LDS stores (transposed layout sB[n][k]).
__device__ __forceinline__ void loadB_f16(const _Float16* __restrict__ B, int K, int N,
                                          int n0, int k0, int tid, v4h* br) {
  const int n4 = (tid & 31) * 4;
  const int kb = (tid >> 5) * 4;
  const bool full = (n0 + TN <= N) && (k0 + TK <= K) && ((N & 3) == 0);
#pragma unroll
  for (int j = 0; j < 4; ++j) {
    int kg = k0 + kb + j, n = n0 + n4;
    if (full) {
      br[j] = *(const v4h*)(B + (size_t)kg * N + n);
    } else {
      int kc = kg < K ? kg : K - 1;
#pragma unroll
      for (int c = 0; c < 4; ++c) {
        int nc = (n + c) < N ? (n + c) : N - 1;
        _Float16 v = B[(size_t)kc * N + nc];
        br[j][c] = (kg < K && (n + c) < N) ? v : (_Float16)0.0f;
      }
    }
  }
}

__device__ __forceinline__ void store_lds(_Float16* sAb, _Float16* sBb, int tid,
                                          const v4f* ar, const v4h* br) {
  const int c4 = (tid & 7) * 4;
#pragma unroll
  for (int i = 0; i < 4; ++i) {
    int r = (tid >> 3) + 32 * i;
    v4h hv;
#pragma unroll
    for (int j = 0; j < 4; ++j) hv[j] = (_Float16)ar[i][j];
    *(v4h*)(sAb + r * LDR + c4) = hv;
  }
  const int n4 = (tid & 31) * 4;
  const int kb = (tid >> 5) * 4;
#pragma unroll
  for (int c = 0; c < 4; ++c) {
    v4h hv;
#pragma unroll
    for (int j = 0; j < 4; ++j) hv[j] = br[j][c];
    *(v4h*)(sBb + (n4 + c) * LDR + kb) = hv;   // 8B-aligned ds_store_b64
  }
}

// ---------------- GEMM: C[M,N] = A[M,K] @ Bf16[K,N] (+bias) (opt relu) -------
// 128x128 block tile, 8 waves, each wave 64x32 (4 M-subtiles x 2 N-subtiles),
// double-buffered LDS, 64 WMMAs per K-chunk per block.
__global__ __launch_bounds__(256) void gemm_wmma(
    const float* __restrict__ A, const _Float16* __restrict__ B,
    const float* __restrict__ bias, float* __restrict__ C,
    int M, int K, int N, int relu)
{
  __shared__ __attribute__((aligned(16))) _Float16 sA[2][TM * LDR];
  __shared__ __attribute__((aligned(16))) _Float16 sB[2][TN * LDR];
  const int tid  = threadIdx.x;
  const int wave = tid >> 5, lane = tid & 31;
  const int lr = lane & 15, lh = lane >> 4;
  const int wr = wave >> 2, wc = wave & 3;
  const int m0 = blockIdx.x * TM;
  const int n0 = blockIdx.y * TN;
  v8f acc[4][2];
#pragma unroll
  for (int q = 0; q < 4; ++q)
#pragma unroll
    for (int p = 0; p < 2; ++p) acc[q][p] = (v8f)0.0f;

  const int Kpad = (K + TK - 1) & ~(TK - 1);
  const int nk = Kpad / TK;
  v4f ar[4];
  v4h br[4];
  loadA_f32(A, M, K, m0, 0, tid, ar);
  loadB_f16(B, K, N, n0, 0, tid, br);
  store_lds(sA[0], sB[0], tid, ar, br);
  __syncthreads();
  for (int kc = 0; kc < nk; ++kc) {
    const int cur = kc & 1;
    const bool more = (kc + 1) < nk;
    if (more) {
      loadA_f32(A, M, K, m0, (kc + 1) * TK, tid, ar);
      loadB_f16(B, K, N, n0, (kc + 1) * TK, tid, br);
    }
    v16h af[4], bf[2];
#pragma unroll
    for (int q = 0; q < 4; ++q) af[q] = frag_ld(&sA[cur][(16 * (4 * wr + q) + lr) * LDR], lh);
#pragma unroll
    for (int p = 0; p < 2; ++p) bf[p] = frag_ld(&sB[cur][(16 * (2 * wc + p) + lr) * LDR], lh);
#pragma unroll
    for (int q = 0; q < 4; ++q)
#pragma unroll
      for (int p = 0; p < 2; ++p)
        acc[q][p] = __builtin_amdgcn_wmma_f32_16x16x32_f16(
            false, af[q], false, bf[p], (short)0, acc[q][p], false, false);
    if (more) store_lds(sA[cur ^ 1], sB[cur ^ 1], tid, ar, br);
    __syncthreads();
  }
  // D layout: VGPR i, lane -> (m = 8*(lane/16)+i, n = lane%16)
#pragma unroll
  for (int q = 0; q < 4; ++q)
#pragma unroll
    for (int p = 0; p < 2; ++p) {
      int n = n0 + 16 * (2 * wc + p) + lr;
      if (n >= N) continue;
      float bv = bias ? bias[n] : 0.f;
#pragma unroll
      for (int i = 0; i < 8; ++i) {
        int m = m0 + 16 * (4 * wr + q) + 8 * lh + i;
        if (m < M) {
          float v = acc[q][p][i] + bv;
          if (relu) v = fmaxf(v, 0.f);
          C[(size_t)m * N + n] = v;
        }
      }
    }
}

// ---------------- Implicit-GEMM conv (NHWC, f16 HWIO weights, stride 1) ------
__global__ __launch_bounds__(256) void conv_wmma(
    const float* __restrict__ in, const _Float16* __restrict__ w,
    const float* __restrict__ bias, float* __restrict__ out,
    int Nb, int H, int W, int Cin, int Cout,
    int kh, int kw, int padh, int padw, int OH, int OW, int relu)
{
  __shared__ __attribute__((aligned(16))) _Float16 sA[2][TM * LDR];
  __shared__ __attribute__((aligned(16))) _Float16 sB[2][TN * LDR];
  const int tid  = threadIdx.x;
  const int wave = tid >> 5, lane = tid & 31;
  const int lr = lane & 15, lh = lane >> 4;
  const int wr = wave >> 2, wc = wave & 3;
  const int M     = Nb * OH * OW;
  const int Kdim  = kh * kw * Cin;
  const int kwCin = kw * Cin;
  const int OHOW  = OH * OW;
  const int m0 = blockIdx.x * TM;
  const int n0 = blockIdx.y * TN;
  v8f acc[4][2];
#pragma unroll
  for (int q = 0; q < 4; ++q)
#pragma unroll
    for (int p = 0; p < 2; ++p) acc[q][p] = (v8f)0.0f;

  const int Kpad = (Kdim + TK - 1) & ~(TK - 1);
  const int nk = Kpad / TK;

  if ((Cin & 3) == 0) {
    // -------- vector path: K-invariant row decomposition hoisted --------
    int rb[4], roy[4], rox[4], rok[4];
    const int c4 = (tid & 7) * 4;
#pragma unroll
    for (int i = 0; i < 4; ++i) {
      int r = (tid >> 3) + 32 * i;
      int p = m0 + r;
      rok[i] = (p < M);
      int pc = rok[i] ? p : 0;
      int b = pc / OHOW, rem = pc - (pc / OHOW) * OHOW;
      roy[i] = rem / OW;
      rox[i] = rem - (rem / OW) * OW;
      rb[i]  = b;
    }
    v4f ar[4];
    v4h br[4];
    auto loadA = [&](int k0) {
      int kk = k0 + c4;                      // same kk for all 4 rows
      int dy = kk / kwCin;
      int r2 = kk - dy * kwCin;
      int dx = r2 / Cin;
      int ci = r2 - dx * Cin;
      bool kin = kk < Kdim;                  // Cin%4==0 => kk..kk+3 same pixel
#pragma unroll
      for (int i = 0; i < 4; ++i) {
        int iy = roy[i] + dy - padh, ix = rox[i] + dx - padw;
        bool ok = rok[i] && kin && iy >= 0 && iy < H && ix >= 0 && ix < W;
        size_t idx = ok ? ((((size_t)rb[i] * H + iy) * W + ix) * Cin + ci) : 0;
        v4f v = *(const v4f*)(in + idx);
        ar[i] = ok ? v : (v4f)0.0f;
      }
    };
    loadA(0);
    loadB_f16(w, Kdim, Cout, n0, 0, tid, br);
    store_lds(sA[0], sB[0], tid, ar, br);
    __syncthreads();
    for (int kc = 0; kc < nk; ++kc) {
      const int cur = kc & 1;
      const bool more = (kc + 1) < nk;
      if (more) {
        loadA((kc + 1) * TK);
        loadB_f16(w, Kdim, Cout, n0, (kc + 1) * TK, tid, br);
      }
      v16h af[4], bf[2];
#pragma unroll
      for (int q = 0; q < 4; ++q) af[q] = frag_ld(&sA[cur][(16 * (4 * wr + q) + lr) * LDR], lh);
#pragma unroll
      for (int p = 0; p < 2; ++p) bf[p] = frag_ld(&sB[cur][(16 * (2 * wc + p) + lr) * LDR], lh);
#pragma unroll
      for (int q = 0; q < 4; ++q)
#pragma unroll
        for (int p = 0; p < 2; ++p)
          acc[q][p] = __builtin_amdgcn_wmma_f32_16x16x32_f16(
              false, af[q], false, bf[p], (short)0, acc[q][p], false, false);
      if (more) store_lds(sA[cur ^ 1], sB[cur ^ 1], tid, ar, br);
      __syncthreads();
    }
  } else {
    // -------- scalar fallback (conv1: Cin=3, Kdim=27 -> single chunk) --------
    for (int k0 = 0; k0 < Kpad; k0 += TK) {
      __syncthreads();
#pragma unroll
      for (int i = 0; i < (TM * TK) / 256; ++i) {
        int e = tid + i * 256;
        int r = e >> 5, c = e & 31;
        int p = m0 + r, kk = k0 + c;
        bool pin = p < M;
        int pc = pin ? p : 0;
        int b = pc / OHOW, rem = pc - (pc / OHOW) * OHOW;
        int oy = rem / OW, ox = rem - (rem / OW) * OW;
        int kc2 = kk < Kdim ? kk : 0;
        int dy = kc2 / kwCin, r2 = kc2 - (kc2 / kwCin) * kwCin;
        int dx = r2 / Cin, ci = r2 - (r2 / Cin) * Cin;
        int iy = oy + dy - padh, ix = ox + dx - padw;
        bool ok = pin && kk < Kdim && iy >= 0 && iy < H && ix >= 0 && ix < W;
        size_t idx = ok ? ((((size_t)b * H + iy) * W + ix) * Cin + ci) : 0;
        float v = in[idx];
        sA[0][r * LDR + c] = (_Float16)(ok ? v : 0.f);
      }
#pragma unroll
      for (int i = 0; i < (TN * TK) / 256; ++i) {
        int e = tid + i * 256;
        int k = e >> 7, n = e & 127;
        int kg = k0 + k;
        int kc2 = kg < Kdim ? kg : Kdim - 1;
        int nc  = (n0 + n) < Cout ? (n0 + n) : Cout - 1;
        _Float16 v = w[(size_t)kc2 * Cout + nc];
        sB[0][n * LDR + k] = (kg < Kdim && (n0 + n) < Cout) ? v : (_Float16)0.0f;
      }
      __syncthreads();
      v16h af[4], bf[2];
#pragma unroll
      for (int q = 0; q < 4; ++q) af[q] = frag_ld(&sA[0][(16 * (4 * wr + q) + lr) * LDR], lh);
#pragma unroll
      for (int p = 0; p < 2; ++p) bf[p] = frag_ld(&sB[0][(16 * (2 * wc + p) + lr) * LDR], lh);
#pragma unroll
      for (int q = 0; q < 4; ++q)
#pragma unroll
        for (int p = 0; p < 2; ++p)
          acc[q][p] = __builtin_amdgcn_wmma_f32_16x16x32_f16(
              false, af[q], false, bf[p], (short)0, acc[q][p], false, false);
    }
    __syncthreads();
  }

#pragma unroll
  for (int q = 0; q < 4; ++q)
#pragma unroll
    for (int p = 0; p < 2; ++p) {
      int n = n0 + 16 * (2 * wc + p) + lr;
      if (n >= Cout) continue;
      float bv = bias[n];
#pragma unroll
      for (int i = 0; i < 8; ++i) {
        int m = m0 + 16 * (4 * wr + q) + 8 * lh + i;
        if (m < M) {
          float v = acc[q][p][i] + bv;
          if (relu) v = fmaxf(v, 0.f);
          out[(size_t)m * Cout + n] = v;
        }
      }
    }
}

// ---------------- weight pre-conversion f32 -> f16 (once per call) -----------
__global__ void cvt16(const float* __restrict__ x, _Float16* __restrict__ o, int n)
{
  int i = blockIdx.x * 256 + threadIdx.x;
  if (i < n) o[i] = (_Float16)x[i];
}

// ---------------- BatchNorm (train-mode) stats + fused normalize+ReLU --------
__global__ void bn_stats(const float* __restrict__ x, float* __restrict__ mv,
                         int Melem, int C)
{
  __shared__ float ss[256], sq[256];
  int c = blockIdx.x;
  float s = 0.f, q = 0.f;
  for (int i = threadIdx.x; i < Melem; i += 256) {
    float v = x[(size_t)i * C + c];
    s += v; q += v * v;
  }
  ss[threadIdx.x] = s; sq[threadIdx.x] = q;
  __syncthreads();
  for (int o = 128; o > 0; o >>= 1) {
    if (threadIdx.x < o) { ss[threadIdx.x] += ss[threadIdx.x + o];
                           sq[threadIdx.x] += sq[threadIdx.x + o]; }
    __syncthreads();
  }
  if (threadIdx.x == 0) {
    float m = ss[0] / (float)Melem;
    mv[c]     = m;
    mv[C + c] = sq[0] / (float)Melem - m * m;
  }
}

__global__ void bn_relu(float* __restrict__ x, const float* __restrict__ mv,
                        const float* __restrict__ scale, const float* __restrict__ bias,
                        int total, int C)
{
  int i = blockIdx.x * 256 + threadIdx.x;
  if (i >= total) return;
  int c = i % C;
  float y = (x[i] - mv[c]) * rsqrtf(mv[C + c] + 1e-5f) * scale[c] + bias[c];
  x[i] = fmaxf(y, 0.f);
}

// ---------------- 2x2 max pool, NHWC, SAME (pad-low = 0 for our shapes) ------
__global__ void maxpool2(const float* __restrict__ in, float* __restrict__ out,
                         int Nb, int H, int W, int C, int sh, int sw, int OH, int OW)
{
  int i = blockIdx.x * 256 + threadIdx.x;
  int total = Nb * OH * OW * C;
  if (i >= total) return;
  int c = i % C; int t = i / C;
  int ox = t % OW; t /= OW;
  int oy = t % OH; int b = t / OH;
  float m = -INFINITY;
#pragma unroll
  for (int dy = 0; dy < 2; ++dy)
#pragma unroll
    for (int dx = 0; dx < 2; ++dx) {
      int iy = oy * sh + dy, ix = ox * sw + dx;
      if (iy < H && ix < W)
        m = fmaxf(m, in[(((size_t)b * H + iy) * W + ix) * C + c]);
    }
  out[i] = m;
}

// ---------------- tiny self-attention (one block per batch, T=15, D=512) -----
__global__ void attention(const float* __restrict__ q, const float* __restrict__ k,
                          const float* __restrict__ v, float* __restrict__ out,
                          int T, int D)
{
  __shared__ float sc[16 * 16];   // scores [t][s]
  int b = blockIdx.x;
  const float* qb = q + (size_t)b * T * D;
  const float* kb = k + (size_t)b * T * D;
  const float* vb = v + (size_t)b * T * D;
  float*       ob = out + (size_t)b * T * D;
  for (int e = threadIdx.x; e < T * T; e += blockDim.x) {
    int t = e / T, s = e - (e / T) * T;
    float d = 0.f;
    for (int i = 0; i < D; ++i) d += qb[t * D + i] * kb[s * D + i];
    sc[t * 16 + s] = d * (1.f / 32.f);
  }
  __syncthreads();
  if (threadIdx.x < (unsigned)T) {
    int t = threadIdx.x;
    float mx = -INFINITY;
    for (int s = 0; s < T; ++s) mx = fmaxf(mx, sc[t * 16 + s]);
    float sum = 0.f;
    for (int s = 0; s < T; ++s) { float e = __expf(sc[t * 16 + s] - mx); sc[t * 16 + s] = e; sum += e; }
    float inv = 1.f / sum;
    for (int s = 0; s < T; ++s) sc[t * 16 + s] *= inv;
  }
  __syncthreads();
  // out[b,s,k] = sum_t v[b,t,k] * p[t][s]
  for (int e = threadIdx.x; e < T * D; e += blockDim.x) {
    int s = e / D, kk = e - (e / D) * D;
    float d = 0.f;
    for (int t = 0; t < T; ++t) d += vb[t * D + kk] * sc[t * 16 + s];
    ob[s * D + kk] = d;
  }
}

// ---------------- LSTM gate fusion (flax order i,f,g,o); writes concat slot --
__global__ void lstm_gate(const float* __restrict__ xw, const float* __restrict__ g,
                          float* __restrict__ c, float* __restrict__ h,
                          float* __restrict__ y, int t, int T, int B, int Hd,
                          int ystride, int yoff)
{
  int i = blockIdx.x * 256 + threadIdx.x;
  if (i >= B * Hd) return;
  int b = i / Hd, j = i - (i / Hd) * Hd;
  size_t xr = ((size_t)b * T + t) * (4 * Hd);
  size_t gr = (size_t)b * (4 * Hd);
  float gi = xw[xr + j]          + g[gr + j];
  float gf = xw[xr + Hd + j]     + g[gr + Hd + j];
  float gg = xw[xr + 2 * Hd + j] + g[gr + 2 * Hd + j];
  float go = xw[xr + 3 * Hd + j] + g[gr + 3 * Hd + j];
  float si = 1.f / (1.f + __expf(-gi));
  float sf = 1.f / (1.f + __expf(-gf));
  float so = 1.f / (1.f + __expf(-go));
  float cn = sf * c[i] + si * tanhf(gg);
  float hn = so * tanhf(cn);
  c[i] = cn; h[i] = hn;
  y[((size_t)b * T + t) * ystride + yoff + j] = hn;
}

__global__ void fill0(float* __restrict__ p, int n)
{
  int i = blockIdx.x * 256 + threadIdx.x;
  if (i < n) p[i] = 0.f;
}

static inline int cdiv(int a, int b) { return (a + b - 1) / b; }

extern "C" void kernel_launch(void* const* d_in, const int* in_sizes, int n_in,
                              void* d_out, int out_size, void* d_ws, size_t ws_size,
                              hipStream_t stream)
{
  const float* x   = (const float*)d_in[0];
  const float* c1b = (const float*)d_in[2];
  const float* c2b = (const float*)d_in[4];
  const float* c3b = (const float*)d_in[6];
  const float* b1s = (const float*)d_in[7];  const float* b1b = (const float*)d_in[8];
  const float* c4b = (const float*)d_in[10];
  const float* b2s = (const float*)d_in[11]; const float* b2b = (const float*)d_in[12];
  const float* c5b = (const float*)d_in[14];
  const float* b3s = (const float*)d_in[15]; const float* b3b = (const float*)d_in[16];
  const float* c6b = (const float*)d_in[18];
  const float* b4s = (const float*)d_in[19]; const float* b4b = (const float*)d_in[20];
  const float* c7b = (const float*)d_in[22];
  const float* b5s = (const float*)d_in[23]; const float* b5b = (const float*)d_in[24];
  const float* bq = (const float*)d_in[26];
  const float* bk = (const float*)d_in[28];
  const float* bv = (const float*)d_in[30];
  const float* l1f_b = (const float*)d_in[33];
  const float* l1r_b = (const float*)d_in[36];
  const float* l2f_b = (const float*)d_in[39];
  const float* l2r_b = (const float*)d_in[42];
  const float* bo = (const float*)d_in[44];
  float* out = (float*)d_out;

  // ---- workspace carve: ping-pong (fits in 192MB L2) + small buffers ----
  char* ws = (char*)d_ws;
  float* bufA = (float*)ws;                       // >= 134,217,728 B (conv1 out)
  float* bufB = (float*)(ws + 134217728);         // >=  67,108,864 B (conv2 out)
  size_t off = 134217728 + 67108864;
  auto alloc = [&](size_t bytes) -> void* {
    void* p = (void*)(ws + off);
    off += (bytes + 255) & ~(size_t)255;
    return p;
  };
  float* mv   = (float*)alloc(2 * 512 * sizeof(float));
  float* qb   = (float*)alloc(480 * 512 * sizeof(float));
  float* kb   = (float*)alloc(480 * 512 * sizeof(float));
  float* vb   = (float*)alloc(480 * 512 * sizeof(float));
  float* attn = (float*)alloc(480 * 512 * sizeof(float));
  float* xw   = (float*)alloc(480 * 2048 * sizeof(float));
  float* y1   = (float*)alloc(480 * 512 * sizeof(float));
  float* y2   = (float*)alloc(480 * 1024 * sizeof(float));
  float* cb   = (float*)alloc(32 * 512 * sizeof(float));
  float* hb   = (float*)alloc(32 * 512 * sizeof(float));
  float* gb   = (float*)alloc(32 * 2048 * sizeof(float));
  (void)ws_size; (void)n_in; (void)out_size;

  dim3 blk(256);

  // ---- pre-convert all weight matrices to f16 (once per call, ~48MB) ----
  static const int widx[19] = {1, 3, 5, 9, 13, 17, 21, 25, 27, 29,
                               31, 32, 34, 35, 37, 38, 40, 41, 43};
  _Float16* wf[45] = {};
  for (int ii = 0; ii < 19; ++ii) {
    int id = widx[ii];
    int n = in_sizes[id];
    wf[id] = (_Float16*)alloc((size_t)n * sizeof(_Float16));
    cvt16<<<cdiv(n, 256), blk, 0, stream>>>((const float*)d_in[id], wf[id], n);
  }

  const int B = 32;

  auto conv = [&](const float* in, const _Float16* w, const float* bias, float* o,
                  int H, int W, int Cin, int Cout, int kh, int kw,
                  int pad, int OH, int OW, int relu) {
    dim3 grid(cdiv(B * OH * OW, TM), cdiv(Cout, TN));
    conv_wmma<<<grid, blk, 0, stream>>>(in, w, bias, o, B, H, W, Cin, Cout,
                                        kh, kw, pad, pad, OH, OW, relu);
  };
  auto gemm = [&](const float* A, const _Float16* Bm, const float* bias, float* C,
                  int M, int K, int N, int relu) {
    dim3 grid(cdiv(M, TM), cdiv(N, TN));
    gemm_wmma<<<grid, blk, 0, stream>>>(A, Bm, bias, C, M, K, N, relu);
  };
  auto bn = [&](float* buf, int Melem, int C, const float* s, const float* b) {
    bn_stats<<<C, blk, 0, stream>>>(buf, mv, Melem, C);
    bn_relu<<<cdiv(Melem * C, 256), blk, 0, stream>>>(buf, mv, s, b, Melem * C, C);
  };
  auto pool = [&](const float* in, float* o, int H, int W, int C, int sh, int sw,
                  int OH, int OW) {
    maxpool2<<<cdiv(B * OH * OW * C, 256), blk, 0, stream>>>(in, o, B, H, W, C, sh, sw, OH, OW);
  };

  // ---- conv stack ----
  conv(x,    wf[1],  c1b, bufA, 64, 256,   3,  64, 3, 3, 1, 64, 256, 1);  // relu
  pool(bufA, bufB, 64, 256, 64, 2, 2, 32, 128);
  conv(bufB, wf[3],  c2b, bufA, 32, 128,  64, 128, 3, 3, 1, 32, 128, 1);  // relu
  pool(bufA, bufB, 32, 128, 128, 2, 2, 16, 64);
  conv(bufB, wf[5],  c3b, bufA, 16,  64, 128, 256, 3, 3, 1, 16, 64, 0);
  bn(bufA, B * 16 * 64, 256, b1s, b1b);
  conv(bufA, wf[9],  c4b, bufB, 16,  64, 256, 256, 3, 3, 1, 16, 64, 0);
  bn(bufB, B * 16 * 64, 256, b2s, b2b);
  pool(bufB, bufA, 16, 64, 256, 1, 2, 16, 32);
  conv(bufA, wf[13], c5b, bufB, 16,  32, 256, 512, 3, 3, 1, 16, 32, 0);
  bn(bufB, B * 16 * 32, 512, b3s, b3b);
  conv(bufB, wf[17], c6b, bufA, 16,  32, 512, 512, 3, 3, 1, 16, 32, 0);
  bn(bufA, B * 16 * 32, 512, b4s, b4b);
  pool(bufA, bufB, 16, 32, 512, 1, 2, 16, 16);
  conv(bufB, wf[21], c7b, bufA, 16,  16, 512, 512, 2, 2, 0, 15, 15, 0);   // VALID
  bn(bufA, B * 15 * 15, 512, b5s, b5b);

  // ---- attention: X = bufA viewed as (480, 7680) ----
  const int T = 15, Dq = 512, Din = 15 * 512, M = B * T;
  gemm(bufA, wf[25], bq, qb, M, Din, Dq, 0);
  gemm(bufA, wf[27], bk, kb, M, Din, Dq, 0);
  gemm(bufA, wf[29], bv, vb, M, Din, Dq, 0);
  attention<<<B, blk, 0, stream>>>(qb, kb, vb, attn, T, Dq);

  // ---- LSTM stack 1 (H=256, bidirectional; reverse continues forward carry) ----
  {
    const int Hd = 256;
    gemm(attn, wf[31], l1f_b, xw, M, 512, 4 * Hd, 0);
    fill0<<<cdiv(B * Hd, 256), blk, 0, stream>>>(cb, B * Hd);
    fill0<<<cdiv(B * Hd, 256), blk, 0, stream>>>(hb, B * Hd);
    for (int t = 0; t < T; ++t) {
      gemm(hb, wf[32], nullptr, gb, B, Hd, 4 * Hd, 0);
      lstm_gate<<<cdiv(B * Hd, 256), blk, 0, stream>>>(xw, gb, cb, hb, y1, t, T, B, Hd, 512, 0);
    }
    gemm(attn, wf[34], l1r_b, xw, M, 512, 4 * Hd, 0);
    for (int t = T - 1; t >= 0; --t) {
      gemm(hb, wf[35], nullptr, gb, B, Hd, 4 * Hd, 0);
      lstm_gate<<<cdiv(B * Hd, 256), blk, 0, stream>>>(xw, gb, cb, hb, y1, t, T, B, Hd, 512, Hd);
    }
  }

  // ---- LSTM stack 2 (H=512); concat order out = [y3(reverse), y2b(forward)] ----
  {
    const int Hd = 512;
    gemm(y1, wf[37], l2f_b, xw, M, 512, 4 * Hd, 0);
    fill0<<<cdiv(B * Hd, 256), blk, 0, stream>>>(cb, B * Hd);
    fill0<<<cdiv(B * Hd, 256), blk, 0, stream>>>(hb, B * Hd);
    for (int t = 0; t < T; ++t) {
      gemm(hb, wf[38], nullptr, gb, B, Hd, 4 * Hd, 0);
      lstm_gate<<<cdiv(B * Hd, 256), blk, 0, stream>>>(xw, gb, cb, hb, y2, t, T, B, Hd, 1024, 512);
    }
    gemm(y1, wf[40], l2r_b, xw, M, 512, 4 * Hd, 0);
    for (int t = T - 1; t >= 0; --t) {
      gemm(hb, wf[41], nullptr, gb, B, Hd, 4 * Hd, 0);
      lstm_gate<<<cdiv(B * Hd, 256), blk, 0, stream>>>(xw, gb, cb, hb, y2, t, T, B, Hd, 1024, 0);
    }
  }

  // ---- output projection (480,1024) @ (1024,1000) + bo -> d_out ----
  gemm(y2, wf[43], bo, out, M, 1024, 1000, 0);
}
